// LiquidReservoir_27616639713790
// MI455X (gfx1250) — matrix operand order repaired and checked by reading, not compiled, not run on hardware
//
#include <hip/hip_runtime.h>
#include <hip/hip_bf16.h>

// ---- problem constants (match reference) ----
#define B_    64
#define S_    512
#define I_    512
#define H_    1024
#define NN_   256
#define KCAT  1536            // I_ + H_
#define LEAK  0.9f
#define ILEAK 0.1f

typedef __attribute__((ext_vector_type(16))) __bf16 v16bf;
typedef __attribute__((ext_vector_type(8)))  float  v8f;

union V16 { v16bf v; uint4 q[2]; };   // 32 bytes: one WMMA 16-bit operand fragment
union F8  { v8f   v; float f[8]; };   // 32 bytes: one WMMA f32 accumulator

// ------------------------------------------------------------------
// Prep kernel (parallel, trivial): build bf16 weights + gather table.
//   Wcat[k][h] : k<512 -> W_in[k][h] ; else W_res[h][k-512]   (B operand of fused GEMM)
//   nwB [h][n] : neuron_w[n][h]                               (B operand of base GEMM)
//   nwT [n][m] : neuron_w[m][idx[n]]                          (rank-1 correction table)
// ------------------------------------------------------------------
__global__ void lr_prep(const float* __restrict__ Win, const float* __restrict__ Wres,
                        const float* __restrict__ nw,
                        __bf16* __restrict__ Wcat, __bf16* __restrict__ nwB,
                        float* __restrict__ nwT) {
  long i = (long)blockIdx.x * blockDim.x + threadIdx.x;
  const long nWcat = (long)KCAT * H_;
  const long nNwB  = (long)H_ * NN_;
  const long nNwT  = (long)NN_ * NN_;
  if (i < nWcat) {
    int k = (int)(i >> 10), h = (int)(i & (H_ - 1));
    float v = (k < I_) ? Win[(long)k * H_ + h] : Wres[(long)h * H_ + (k - I_)];
    Wcat[i] = (__bf16)v;
  } else if (i < nWcat + nNwB) {
    long p = i - nWcat;
    int h = (int)(p >> 8), n = (int)(p & (NN_ - 1));
    nwB[p] = (__bf16)nw[(long)n * H_ + h];
  } else if (i < nWcat + nNwB + nNwT) {
    long p = i - nWcat - nNwB;
    int n = (int)(p >> 8), m = (int)(p & (NN_ - 1));
    int id = (int)((double)n * (1023.0 / 255.0));  // np.linspace(0,1023,256) truncated
    nwT[p] = nw[(long)m * H_ + id];
  }
}

// ------------------------------------------------------------------
// Persistent single-workgroup reservoir kernel.
// 1024 threads = 32 wave32 waves on one WGP; all recurrent state in LDS.
// ------------------------------------------------------------------
__launch_bounds__(1024)
__global__ void lr_reservoir(const float* __restrict__ x, const float* __restrict__ nb,
                             const __bf16* __restrict__ Wcat, const __bf16* __restrict__ nwB,
                             const float* __restrict__ nwT, float* __restrict__ master,
                             float* __restrict__ out) {
  extern __shared__ char smem[];
  __bf16* xbuf = (__bf16*)(smem);                    //  64 x 512  bf16  (65536 B)
  __bf16* sbuf = (__bf16*)(smem + 65536);            //  64 x 1024 bf16  (131072 B)
  float*  wi   = (float*)(smem + 65536 + 131072);    //  64 x 256  f32   (65536 B)
  float*  ns   = (float*)(smem + 262144);            //  256 f32
  int*    idxs = (int*)  (smem + 263168);            //  256 i32

  const int tid  = threadIdx.x;
  const int lane = tid & 31;
  const int wave = tid >> 5;
  const int am   = lane & 15;           // row-in-tile for A / col-in-tile for D
  const int klo  = (lane >> 4) << 3;    // K sub-offset for A fragments
  const int bhi  = (lane >> 4) << 3;    // M offset for C/D fragments

  // ---- init: zero state (f32 master + bf16 shadow) and neuron states ----
  for (int i = tid; i < B_ * H_; i += 1024) { sbuf[i] = (__bf16)0.0f; master[i] = 0.0f; }
  if (tid < NN_) { ns[tid] = 0.0f; idxs[tid] = (int)((double)tid * (1023.0 / 255.0)); }
  __threadfence();
  __syncthreads();

  for (int t = 0; t < S_; ++t) {
    // ---- stage x_t into LDS as bf16 ----
    for (int i = tid; i < B_ * I_; i += 1024) {
      int b = i >> 9, k = i & (I_ - 1);
      xbuf[i] = (__bf16)x[(size_t)b * (S_ * I_) + (size_t)t * I_ + k];
    }
    __syncthreads();

    // ---- phase 1: pre = [x_t | state] @ Wcat  (M=64,N=1024,K=1536), 8 tiles/wave ----
    const int bt   = wave >> 3;
    const int ht0  = (wave & 7) << 3;
    const int arow = bt * 16 + am;
    F8 acc[8];
    #pragma unroll
    for (int j = 0; j < 8; ++j)
      #pragma unroll
      for (int g = 0; g < 8; ++g) acc[j].f[g] = 0.0f;

    for (int k0 = 0; k0 < KCAT; k0 += 32) {
      V16 a;
      const __bf16* ap = (k0 < I_) ? (xbuf + arow * I_ + (k0 + klo))
                                   : (sbuf + arow * H_ + (k0 - I_ + klo));
      a.q[0] = *(const uint4*)(ap);
      a.q[1] = *(const uint4*)(ap + 16);
      const __bf16* brow = Wcat + (size_t)(k0 + klo) * H_ + am;   // lane holds B column am; elements gather K rows in table order
      #pragma unroll
      for (int j = 0; j < 8; ++j) {
        V16 bm;
        #pragma unroll
        for (int s = 0; s < 16; ++s) bm.v[s] = brow[(size_t)(s + (s & 8)) * H_ + ((ht0 + j) << 4)];
        acc[j].v = __builtin_amdgcn_wmma_f32_16x16x32_bf16(
            false, a.v, false, bm.v, (short)0, acc[j].v, false, false);
      }
    }
    __syncthreads();   // all old-state reads done before anyone rewrites state

    // ---- phase 2: state = LEAK*state + (1-LEAK)*tanh(pre); write out/master/LDS ----
    {
      const int brow0 = bt * 16 + bhi;
      #pragma unroll
      for (int j = 0; j < 8; ++j) {
        const int h = ((ht0 + j) << 4) + am;
        #pragma unroll
        for (int g = 0; g < 8; ++g) {
          const int b  = brow0 + g;
          float old    = master[b * H_ + h];
          float sn     = LEAK * old + ILEAK * tanhf(acc[j].f[g]);
          master[b * H_ + h] = sn;
          sbuf[b * H_ + h]   = (__bf16)sn;
          out[(size_t)b * (S_ * H_) + (size_t)t * H_ + h] = sn;
        }
      }
    }
    __syncthreads();

    // ---- phase 3: base wi = state_new @ neuron_w^T + b  (M=64,N=256,K=1024) ----
    {
      const int bt3   = wave >> 3;
      const int nt0   = (wave & 7) << 1;
      const int arow3 = bt3 * 16 + am;
      F8 acc2[2];
      #pragma unroll
      for (int j = 0; j < 2; ++j)
        #pragma unroll
        for (int g = 0; g < 8; ++g) acc2[j].f[g] = 0.0f;

      for (int k0 = 0; k0 < H_; k0 += 32) {
        V16 a;
        const __bf16* ap = sbuf + arow3 * H_ + (k0 + klo);
        a.q[0] = *(const uint4*)(ap);
        a.q[1] = *(const uint4*)(ap + 16);
        const __bf16* brow = nwB + (size_t)(k0 + klo) * NN_ + am;
        #pragma unroll
        for (int j = 0; j < 2; ++j) {
          V16 bm;
          #pragma unroll
          for (int s = 0; s < 16; ++s) bm.v[s] = brow[(size_t)(s + (s & 8)) * NN_ + ((nt0 + j) << 4)];
          acc2[j].v = __builtin_amdgcn_wmma_f32_16x16x32_bf16(
              false, a.v, false, bm.v, (short)0, acc2[j].v, false, false);
        }
      }
      const int brow0 = bt3 * 16 + bhi;
      #pragma unroll
      for (int j = 0; j < 2; ++j) {
        const int n = ((nt0 + j) << 4) + am;
        #pragma unroll
        for (int g = 0; g < 8; ++g) {
          const int b = brow0 + g;
          wi[b * NN_ + n] = acc2[j].f[g] + nb[n];
        }
      }
    }
    __syncthreads();

    // ---- phase 4: serial neuron scan (wave 0 owns the chain; all waves do rank-1) ----
    for (int n = 0; n < NN_; ++n) {
      if (wave == 0) {
        const int b0 = lane, b1 = lane + 32;        // 2 batch rows per lane
        float w0 = wi[b0 * NN_ + n];
        float w1 = wi[b1 * NN_ + n];
        float s  = w0 + w1;
        #pragma unroll
        for (int off = 16; off > 0; off >>= 1) s += __shfl_xor(s, off, 32);
        float nn2 = LEAK * ns[n] + ILEAK * (s * (1.0f / 64.0f));
        ns[n] = nn2;                                 // all lanes write same value
        const int id = idxs[n];
        float o0 = (float)sbuf[b0 * H_ + id];
        float o1 = (float)sbuf[b1 * H_ + id];
        float a0 = tanhf(w0 + nn2);
        float a1 = tanhf(w1 + nn2);
        sbuf[b0 * H_ + id] = (__bf16)a0;
        sbuf[b1 * H_ + id] = (__bf16)a1;
        master[b0 * H_ + id] = a0;
        master[b1 * H_ + id] = a1;
        out[(size_t)b0 * (S_ * H_) + (size_t)t * H_ + id] = a0;
        out[(size_t)b1 * (S_ * H_) + (size_t)t * H_ + id] = a1;
        wi[b0 * NN_ + n] = a0 - o0;                  // stash delta in-place
        wi[b1 * NN_ + n] = a1 - o1;
      }
      __syncthreads();
      {  // lazy rank-1 correction: wi[b][m>n] += nw[m, idx[n]] * delta[b]
        const int b = tid & 63;
        const int c = tid >> 6;
        const float d = wi[b * NN_ + n];
        const float* wrow = nwT + (size_t)n * NN_ + (c << 4);
        #pragma unroll
        for (int j2 = 0; j2 < 16; ++j2) {
          const int m = (c << 4) + j2;
          if (m > n) wi[b * NN_ + m] += wrow[j2] * d;
        }
      }
      __syncthreads();
    }

    __threadfence();   // make scan's master stores visible to phase-2 readers next step
    __syncthreads();
  }
}

// ------------------------------------------------------------------
extern "C" void kernel_launch(void* const* d_in, const int* in_sizes, int n_in,
                              void* d_out, int out_size, void* d_ws, size_t ws_size,
                              hipStream_t stream) {
  (void)in_sizes; (void)n_in; (void)out_size; (void)ws_size;
  const float* x    = (const float*)d_in[0];
  const float* Win  = (const float*)d_in[1];
  const float* Wres = (const float*)d_in[2];
  const float* nw   = (const float*)d_in[3];
  const float* nb   = (const float*)d_in[4];
  float* out = (float*)d_out;

  char* ws = (char*)d_ws;
  __bf16* Wcat   = (__bf16*)(ws);              // 1536*1024*2 = 3,145,728 B
  __bf16* nwB    = (__bf16*)(ws + 3145728);    // 1024*256*2  =   524,288 B
  float*  nwT    = (float*) (ws + 3670016);    // 256*256*4   =   262,144 B
  float*  master = (float*) (ws + 3932160);    // 64*1024*4   =   262,144 B  (4.2 MB total)

  const long total = (long)KCAT * H_ + (long)H_ * NN_ + (long)NN_ * NN_;
  const int blocks = (int)((total + 255) / 256);
  lr_prep<<<blocks, 256, 0, stream>>>(Win, Wres, nw, Wcat, nwB, nwT);

  const size_t lds = 264448;  // xbuf 64K + sbuf 128K + wi 64K + ns/idx/pad
  lr_reservoir<<<1, 1024, lds, stream>>>(x, nb, Wcat, nwB, nwT, master, out);
}
